// TTTLinear_77309412112
// MI455X (gfx1250) — matrix operand order repaired and checked by reading, not compiled, not run on hardware
//
#include <hip/hip_runtime.h>
#include <hip/hip_bf16.h>

typedef float v2f __attribute__((ext_vector_type(2)));
typedef float v8f __attribute__((ext_vector_type(8)));
typedef __bf16 bf16_t;
typedef __bf16 v16bf __attribute__((ext_vector_type(16)));

#define NH_ 32
#define HS_ 2048
#define K_  16
#define HD_ 64
#define B_  4
#define L_  2048
#define NM_ 128
#define M_  (B_ * L_)   // 8192 tokens

static __device__ __forceinline__ v8f v8zero() {
  v8f z;
#pragma unroll
  for (int i = 0; i < 8; ++i) z[i] = 0.0f;
  return z;
}

// fp32 WMMA: D(16x16) = A(16x4) * B(4x16) + C   -> v_wmma_f32_16x16x4_f32
static __device__ __forceinline__ v8f wmma_f32k4(v2f a, v2f b, v8f c) {
  return __builtin_amdgcn_wmma_f32_16x16x4_f32(false, a, false, b, (short)0, c, false, false);
}
// bf16 WMMA: D(16x16 f32) = A(16x32 bf16) * B(32x16 bf16) + C
static __device__ __forceinline__ v8f wmma_bf16k32(v16bf a, v16bf b, v8f c) {
  return __builtin_amdgcn_wmma_f32_16x16x32_bf16(false, a, false, b, (short)0, c, false, false);
}

// A-frag (16x32 bf16): lane<16 holds K 0..15, lane>=16 K 16..31
static __device__ __forceinline__ v16bf load_afrag(const bf16_t* __restrict__ p) {
  v16bf f;
#pragma unroll
  for (int i = 0; i < 16; ++i) f[i] = p[i];
  return f;
}
// B-frag (32x16 bf16): lane = column; lane<16 holds K 0..15, lane>=16 K 16..31
static __device__ __forceinline__ v16bf load_bfrag(const bf16_t* __restrict__ p) {
  v16bf f;
#pragma unroll
  for (int i = 0; i < 16; ++i) f[i] = p[i];
  return f;
}

// Async copy of one contiguous 16x64 f32 tile into padded LDS [16][65]
// (2 x global_load_async_to_lds_b128 per lane, tracked by ASYNCcnt).
static __device__ __forceinline__ void async_tile_to_lds(const float* __restrict__ g,
                                                         unsigned lds_base, int tid) {
#pragma unroll
  for (int half = 0; half < 2; ++half) {
    int idx = tid + half * 128;          // 256 chunks of 16B
    int r = idx >> 4;                    // row 0..15
    int cc = (idx & 15) * 4;             // col 0..60 step 4
    unsigned la = lds_base + (unsigned)((r * 65 + cc) * 4);
    const float* ga = g + idx * 4;
    asm volatile("global_load_async_to_lds_b128 %0, %1, off"
                 :: "v"(la), "v"(ga)
                 : "memory");
  }
}

// ---------------------------------------------------------------- converts
__global__ __launch_bounds__(256) void cvt_bf16_kernel(const float* __restrict__ s,
                                                       bf16_t* __restrict__ d, int n) {
  int i = blockIdx.x * 256 + threadIdx.x;
  if (i < n) d[i] = (bf16_t)s[i];
}

// -------------------------------------------- ttt_lr sigmoid (block/token)
__global__ __launch_bounds__(256) void lr_sig_kernel(const float* __restrict__ X,
                                                     const float* __restrict__ lrw,
                                                     const float* __restrict__ lrb,
                                                     float* __restrict__ SIG) {
  __shared__ float xrow[HS_];
  __shared__ float part[NH_][8];
  int m = blockIdx.x;                   // token 0..8191
  int b = m >> 11, t = m & 2047;
  const float* x = X + (size_t)m * HS_;
  for (int c = threadIdx.x; c < HS_; c += 256) xrow[c] = x[c];
  __syncthreads();
  int h = threadIdx.x >> 3, seg = threadIdx.x & 7;
  const float* w = lrw + (size_t)h * HS_ + seg * 256;
  const float* xs = xrow + seg * 256;
  float s = 0.f;
  for (int i = 0; i < 256; ++i) s += xs[i] * w[i];
  part[h][seg] = s;
  __syncthreads();
  if (threadIdx.x < NH_) {
    float acc = 0.f;
    for (int p = 0; p < 8; ++p) acc += part[threadIdx.x][p];
    acc += lrb[threadIdx.x];
    float sg = 1.f / (1.f + __expf(-acc));
    SIG[(size_t)(b * NH_ + threadIdx.x) * L_ + t] = sg * (1.f / 64.f);
  }
}

// ---------------------------------------------------------- bf16 WMMA GEMM
// Y = A(M x Kd) @ W(N x Kd)^T.  Each wave: 32 rows x 64 cols (2 m-strips).
// Ping-pong unroll-by-2: two independent fragment register sets, each
// refilled by loads one full iteration ahead of its WMMA consumers — no
// cross-buffer register copies.  mode 0: row-major store; mode 1: QKV scatter.
__global__ __launch_bounds__(128) void gemm_bf16_kernel(const bf16_t* __restrict__ A,
                                                        const bf16_t* __restrict__ W,
                                                        float* __restrict__ out,
                                                        int Kd, int N, int mode) {
  int tid = threadIdx.x, lane = tid & 31, wv = tid >> 5;
  int col = lane & 15, hi = lane >> 4;
  int m0 = blockIdx.x * 128 + wv * 16;
  int n0 = blockIdx.y * 64;

  v8f acc[2][4];
#pragma unroll
  for (int u = 0; u < 2; ++u)
#pragma unroll
    for (int cb = 0; cb < 4; ++cb) acc[u][cb] = v8zero();

  const bf16_t* ar0 = A + (size_t)(m0 + col) * Kd + hi * 16;
  const bf16_t* ar1 = A + (size_t)(m0 + 64 + col) * Kd + hi * 16;
  const bf16_t* br[4];
#pragma unroll
  for (int cb = 0; cb < 4; ++cb) br[cb] = W + (size_t)(n0 + cb * 16 + col) * Kd + hi * 16;

  // two register fragment sets (ping-pong)
  v16bf a0p0, a1p0, bp0[4];
  v16bf a0p1, a1p1, bp1[4];
  a0p0 = load_afrag(ar0);
  a1p0 = load_afrag(ar1);
#pragma unroll
  for (int cb = 0; cb < 4; ++cb) bp0[cb] = load_bfrag(br[cb]);
  a0p1 = load_afrag(ar0 + 32);
  a1p1 = load_afrag(ar1 + 32);
#pragma unroll
  for (int cb = 0; cb < 4; ++cb) bp1[cb] = load_bfrag(br[cb] + 32);

  int k0 = 0;
  for (; k0 < Kd - 64; k0 += 64) {
    // phase 0: consume set0 (loads for it were issued a full iteration ago)
#pragma unroll
    for (int cb = 0; cb < 4; ++cb) {
      acc[0][cb] = wmma_bf16k32(a0p0, bp0[cb], acc[0][cb]);
      acc[1][cb] = wmma_bf16k32(a1p0, bp0[cb], acc[1][cb]);
    }
    // refill set0 with k0+64
    a0p0 = load_afrag(ar0 + k0 + 64);
    a1p0 = load_afrag(ar1 + k0 + 64);
#pragma unroll
    for (int cb = 0; cb < 4; ++cb) bp0[cb] = load_bfrag(br[cb] + k0 + 64);
    __builtin_prefetch(ar0 + k0 + 576, 0, 1);
    __builtin_prefetch(ar1 + k0 + 576, 0, 1);
    // phase 1: consume set1
#pragma unroll
    for (int cb = 0; cb < 4; ++cb) {
      acc[0][cb] = wmma_bf16k32(a0p1, bp1[cb], acc[0][cb]);
      acc[1][cb] = wmma_bf16k32(a1p1, bp1[cb], acc[1][cb]);
    }
    // refill set1 with k0+96
    a0p1 = load_afrag(ar0 + k0 + 96);
    a1p1 = load_afrag(ar1 + k0 + 96);
#pragma unroll
    for (int cb = 0; cb < 4; ++cb) bp1[cb] = load_bfrag(br[cb] + k0 + 96);
  }
  // epilogue: last two phases, no refill
#pragma unroll
  for (int cb = 0; cb < 4; ++cb) {
    acc[0][cb] = wmma_bf16k32(a0p0, bp0[cb], acc[0][cb]);
    acc[1][cb] = wmma_bf16k32(a1p0, bp0[cb], acc[1][cb]);
  }
#pragma unroll
  for (int cb = 0; cb < 4; ++cb) {
    acc[0][cb] = wmma_bf16k32(a0p1, bp1[cb], acc[0][cb]);
    acc[1][cb] = wmma_bf16k32(a1p1, bp1[cb], acc[1][cb]);
  }

#pragma unroll
  for (int u = 0; u < 2; ++u) {
#pragma unroll
    for (int cb = 0; cb < 4; ++cb) {
      int n = n0 + cb * 16 + col;
#pragma unroll
      for (int r = 0; r < 8; ++r) {
        int m = m0 + u * 64 + r + hi * 8;
        float v = acc[u][cb][r];
        if (mode == 0) {
          out[(size_t)m * N + n] = v;
        } else {
          int b = m >> 11, t = m & 2047;
          int h = n >> 6,  d = n & 63;
          out[(((size_t)(b * NH_ + h) * NM_ + (t >> 4)) * K_ + (t & 15)) * HD_ + d] = v;
        }
      }
    }
  }
}

// --------------------------------------------------------------- TTT scan
// 1 WG per (b,h): 4 waves, each owns a 16-col slab of the 64-wide state.
// W1/b1 state lives in LDS; small matmuls via v_wmma_f32_16x16x4_f32.
// xq/xk/xv tiles double-buffered in LDS, filled by GLOBAL_LOAD_ASYNC_TO_LDS
// one step ahead (s_wait_asynccnt pipelining).
__global__ __launch_bounds__(128) void ttt_scan_kernel(const float* __restrict__ XQ,
                                                       const float* __restrict__ XK,
                                                       const float* __restrict__ XV,
                                                       const float* __restrict__ SIG,
                                                       const float* __restrict__ W1g,
                                                       const float* __restrict__ b1g,
                                                       const float* __restrict__ lnw,
                                                       const float* __restrict__ lnb,
                                                       const float* __restrict__ tklearn,
                                                       float* __restrict__ O) {
  __shared__ float Wl[64][65];                        // W1 state
  __shared__ float xqs[2][16][65], xks[2][16][65], xvs[2][16][65];
  __shared__ float zbs[16][65];                       // Z1 then Z1_bar
  __shared__ float gbs[16][65];                       // grad
  __shared__ float gxs[16][65];                       // gxh scratch
  __shared__ float tbs[16][17];                       // eta*tril(Attn)
  __shared__ float redA[16][8], redB[16][8];
  __shared__ float mu_s[16], rstd_s[16], sga_s[16], sgb_s[16];
  __shared__ float sig_s[16], tok_s[16], le_s[16];
  __shared__ float b1_s[64], gam_s[64], bet_s[64];

  const int bh = blockIdx.x;
  const int b = bh >> 5, h = bh & 31;
  const int tid = threadIdx.x;
  const int lane = tid & 31, wv = tid >> 5;
  const int col = lane & 15, hi = lane >> 4;
  const int wN = wv * 16;
  const int prow = tid >> 3, pseg = tid & 7;  // 16 rows x 8 segments

  for (int i = tid; i < 64 * 64; i += 128) Wl[i >> 6][i & 63] = W1g[(size_t)h * 4096 + i];
  if (tid < 64) {
    b1_s[tid]  = b1g[h * 64 + tid];
    gam_s[tid] = lnw[h * 64 + tid];
    bet_s[tid] = lnb[h * 64 + tid];
  }
  if (tid < 16) {
    float tv = 1.f / (float)(tid + 1) + tklearn[tid];
    tok_s[tid] = tv > 0.f ? tv : 0.f;
  }

  const size_t base = (size_t)bh * (NM_ * K_ * HD_);
  const float* sigptr = SIG + (size_t)bh * (NM_ * K_);

  // kick off async fill of buffer 0 (step nm=0)
  async_tile_to_lds(XQ + base, (unsigned)(size_t)&xqs[0][0][0], tid);
  async_tile_to_lds(XK + base, (unsigned)(size_t)&xks[0][0][0], tid);
  async_tile_to_lds(XV + base, (unsigned)(size_t)&xvs[0][0][0], tid);

  for (int nm = 0; nm < NM_; ++nm) {
    const int cur = nm & 1;
    if (nm + 1 < NM_) {
      const size_t nb = base + (size_t)(nm + 1) * 1024;
      async_tile_to_lds(XQ + nb, (unsigned)(size_t)&xqs[cur ^ 1][0][0], tid);
      async_tile_to_lds(XK + nb, (unsigned)(size_t)&xks[cur ^ 1][0][0], tid);
      async_tile_to_lds(XV + nb, (unsigned)(size_t)&xvs[cur ^ 1][0][0], tid);
      asm volatile("s_wait_asynccnt 6" ::: "memory");  // step-nm tiles resident
    } else {
      asm volatile("s_wait_asynccnt 0" ::: "memory");
    }
    if (tid < 16) {
      float s = sigptr[nm * K_ + tid];
      sig_s[tid] = s;
      le_s[tid]  = tok_s[15] * s;   // last_eta
    }
    __syncthreads();

    float (&xq)[16][65] = xqs[cur];
    float (&xk)[16][65] = xks[cur];
    float (&xv)[16][65] = xvs[cur];

    // ---- Z1 = xk @ W1 + b1 (wave slab) ----
    v8f accZ = v8zero();
#pragma unroll
    for (int f = 0; f < 16; ++f) {
      int kb = 4 * f + hi * 2;
      v2f a, bb;
      a.x  = xk[col][kb];         a.y  = xk[col][kb + 1];
      bb.x = Wl[kb][wN + col];    bb.y = Wl[kb + 1][wN + col];
      accZ = wmma_f32k4(a, bb, accZ);
    }
#pragma unroll
    for (int r = 0; r < 8; ++r) zbs[r + hi * 8][wN + col] = accZ[r] + b1_s[wN + col];
    __syncthreads();

    // ---- grad = ln_fused_l2_bwd(Z1, xv - xk, gamma, beta) ----
    {
      float s = 0.f, s2 = 0.f;
#pragma unroll
      for (int c0 = 0; c0 < 8; ++c0) {
        float zv = zbs[prow][pseg * 8 + c0];
        s += zv; s2 += zv * zv;
      }
      redA[prow][pseg] = s; redB[prow][pseg] = s2;
    }
    __syncthreads();
    if (tid < 16) {
      float s = 0.f, s2 = 0.f;
      for (int p = 0; p < 8; ++p) { s += redA[tid][p]; s2 += redB[tid][p]; }
      float m = s * (1.f / 64.f);
      float var = s2 * (1.f / 64.f) - m * m;
      mu_s[tid] = m;
      rstd_s[tid] = 1.f / sqrtf(var + 1e-6f);
    }
    __syncthreads();
    {
      float s1 = 0.f, s2 = 0.f;
      float m = mu_s[prow], rs = rstd_s[prow];
#pragma unroll
      for (int c0 = 0; c0 < 8; ++c0) {
        int c = pseg * 8 + c0;
        float xh = (zbs[prow][c] - m) * rs;
        float tgt = xv[prow][c] - xk[prow][c];
        float g = (gam_s[c] * xh + bet_s[c] - tgt) * gam_s[c];
        gxs[prow][c] = g;
        s1 += g; s2 += g * xh;
      }
      redA[prow][pseg] = s1; redB[prow][pseg] = s2;
    }
    __syncthreads();
    if (tid < 16) {
      float a = 0.f, c2 = 0.f;
      for (int p = 0; p < 8; ++p) { a += redA[tid][p]; c2 += redB[tid][p]; }
      sga_s[tid] = a; sgb_s[tid] = c2;
    }
    __syncthreads();
    {
      float m = mu_s[prow], rs = rstd_s[prow];
      float sa = sga_s[prow], sb = sgb_s[prow];
#pragma unroll
      for (int c0 = 0; c0 < 8; ++c0) {
        int c = pseg * 8 + c0;
        float xh = (zbs[prow][c] - m) * rs;
        float g = gxs[prow][c];
        gbs[prow][c] = (64.f * g - sa - xh * sb) * (rs * (1.f / 64.f));
      }
    }
    __syncthreads();

    // ---- Attn = xq @ xk^T (16x16, K=HD) ----
    v8f accA = v8zero();
#pragma unroll
    for (int f = 0; f < 16; ++f) {
      int kb = 4 * f + hi * 2;
      v2f a, bb;
      a.x  = xq[col][kb];  a.y  = xq[col][kb + 1];
      bb.x = xk[col][kb];  bb.y = xk[col][kb + 1];   // B = xk^T
      accA = wmma_f32k4(a, bb, accA);
    }
    if (wv == 0) {
#pragma unroll
      for (int r = 0; r < 8; ++r) {
        int i = r + hi * 8, j = col;
        tbs[i][j] = (i >= j) ? tok_s[i] * sig_s[j] * accA[r] : 0.f;  // eta*tril(Attn)
      }
    }
    __syncthreads();

    // ---- accU = tril(eta)@grad ;  Z1_bar = xq@W1 - T@grad + b1 - accU ----
    v8f accU = v8zero();
#pragma unroll
    for (int f = 0; f < 4; ++f) {
      int kb = 4 * f + hi * 2;
      v2f a, bb;
      float tm = tok_s[col];
      a.x = (col >= kb)     ? tm * sig_s[kb]     : 0.f;  // analytic tril(eta)
      a.y = (col >= kb + 1) ? tm * sig_s[kb + 1] : 0.f;
      bb.x = gbs[kb][wN + col]; bb.y = gbs[kb + 1][wN + col];
      accU = wmma_f32k4(a, bb, accU);
    }
    v8f accZb = v8zero();
#pragma unroll
    for (int f = 0; f < 16; ++f) {
      int kb = 4 * f + hi * 2;
      v2f a, bb;
      a.x  = xq[col][kb];       a.y  = xq[col][kb + 1];
      bb.x = Wl[kb][wN + col];  bb.y = Wl[kb + 1][wN + col];
      accZb = wmma_f32k4(a, bb, accZb);
    }
#pragma unroll
    for (int f = 0; f < 4; ++f) {
      int kb = 4 * f + hi * 2;
      v2f a, bb;
      a.x = -tbs[col][kb]; a.y = -tbs[col][kb + 1];
      bb.x = gbs[kb][wN + col]; bb.y = gbs[kb + 1][wN + col];
      accZb = wmma_f32k4(a, bb, accZb);
    }
#pragma unroll
    for (int r = 0; r < 8; ++r) {
      int i = r + hi * 8, j = wN + col;
      zbs[i][j] = accZb[r] + b1_s[j] - accU[r];
    }
    __syncthreads();

    // ---- out = xq + LN(Z1_bar) ----
    {
      float s = 0.f, s2 = 0.f;
#pragma unroll
      for (int c0 = 0; c0 < 8; ++c0) {
        float zv = zbs[prow][pseg * 8 + c0];
        s += zv; s2 += zv * zv;
      }
      redA[prow][pseg] = s; redB[prow][pseg] = s2;
    }
    __syncthreads();
    if (tid < 16) {
      float s = 0.f, s2 = 0.f;
      for (int p = 0; p < 8; ++p) { s += redA[tid][p]; s2 += redB[tid][p]; }
      float m = s * (1.f / 64.f);
      float var = s2 * (1.f / 64.f) - m * m;
      mu_s[tid] = m;
      rstd_s[tid] = 1.f / sqrtf(var + 1e-6f);
    }
    __syncthreads();
    {
      float m = mu_s[prow], rs = rstd_s[prow];
      float* orow = O + ((size_t)b * L_ + (size_t)nm * K_ + prow) * HS_ + h * HD_;
#pragma unroll
      for (int c0 = 0; c0 < 8; ++c0) {
        int c = pseg * 8 + c0;
        float xh = (zbs[prow][c] - m) * rs;
        orow[c] = xq[prow][c] + gam_s[c] * xh + bet_s[c];
      }
    }
    __syncthreads();

    // ---- state update: b1 -= sum_i le[i]*grad[i,:],  W1 -= (le*xk)^T @ grad ----
    if (tid < 64) {
      float s = 0.f;
      for (int i = 0; i < 16; ++i) s += le_s[i] * gbs[i][tid];
      b1_s[tid] -= s;
    }
#pragma unroll
    for (int mtile = 0; mtile < 4; ++mtile) {
      int m0 = mtile * 16;
      v8f aw;  // seed accumulator with current W1 block (D layout)
#pragma unroll
      for (int r = 0; r < 8; ++r) aw[r] = Wl[m0 + r + hi * 8][wN + col];
#pragma unroll
      for (int f = 0; f < 4; ++f) {
        int kb = 4 * f + hi * 2;
        v2f a, bb;
        a.x = -le_s[kb]     * xk[kb][m0 + col];        // A = -(le*xk)^T
        a.y = -le_s[kb + 1] * xk[kb + 1][m0 + col];
        bb.x = gbs[kb][wN + col]; bb.y = gbs[kb + 1][wN + col];
        aw = wmma_f32k4(a, bb, aw);
      }
#pragma unroll
      for (int r = 0; r < 8; ++r) Wl[m0 + r + hi * 8][wN + col] = aw[r];
    }
    __syncthreads();
  }
}

// -------------------------------------------------------------- post-norm
__global__ __launch_bounds__(256) void postln_kernel(const float* __restrict__ Oin,
                                                     const float* __restrict__ w,
                                                     const float* __restrict__ bta,
                                                     bf16_t* __restrict__ out) {
  __shared__ float rA[256], rB[256];
  __shared__ float smu, srs;
  int row = blockIdx.x;
  const float* x = Oin + (size_t)row * HS_;
  float s = 0.f, s2 = 0.f;
  for (int c = threadIdx.x; c < HS_; c += 256) { float v = x[c]; s += v; s2 += v * v; }
  rA[threadIdx.x] = s; rB[threadIdx.x] = s2;
  __syncthreads();
  for (int st = 128; st > 0; st >>= 1) {
    if (threadIdx.x < st) { rA[threadIdx.x] += rA[threadIdx.x + st]; rB[threadIdx.x] += rB[threadIdx.x + st]; }
    __syncthreads();
  }
  if (threadIdx.x == 0) {
    float m = rA[0] * (1.f / (float)HS_);
    float var = rB[0] * (1.f / (float)HS_) - m * m;
    smu = m; srs = 1.f / sqrtf(var + 1e-6f);
  }
  __syncthreads();
  float m = smu, rs = srs;
  bf16_t* orow = out + (size_t)row * HS_;
  for (int c = threadIdx.x; c < HS_; c += 256)
    orow[c] = (bf16_t)(((x[c] - m) * rs) * w[c] + bta[c]);
}

// ----------------------------------------------------------------- driver
extern "C" void kernel_launch(void* const* d_in, const int* in_sizes, int n_in,
                              void* d_out, int out_size, void* d_ws, size_t ws_size,
                              hipStream_t stream) {
  (void)in_sizes; (void)n_in; (void)out_size; (void)ws_size;
  const float* hs  = (const float*)d_in[0];
  const float* qw  = (const float*)d_in[2];
  const float* kw  = (const float*)d_in[3];
  const float* vw  = (const float*)d_in[4];
  const float* ow  = (const float*)d_in[5];
  const float* W1  = (const float*)d_in[6];
  const float* b1  = (const float*)d_in[7];
  const float* lnw = (const float*)d_in[8];
  const float* lnb = (const float*)d_in[9];
  const float* lrw = (const float*)d_in[10];
  const float* lrb = (const float*)d_in[11];
  const float* tkl = (const float*)d_in[12];
  const float* pnw = (const float*)d_in[13];
  const float* pnb = (const float*)d_in[14];

  char* ws = (char*)d_ws;
  size_t off = 0;
  auto take = [&](size_t bytes) { void* p = ws + off; off += (bytes + 255) & ~(size_t)255; return p; };

  bf16_t* Xbf  = (bf16_t*)take((size_t)M_ * HS_ * 2);
  bf16_t* Wqb  = (bf16_t*)take((size_t)HS_ * HS_ * 2);
  bf16_t* Wkb  = (bf16_t*)take((size_t)HS_ * HS_ * 2);
  bf16_t* Wvb  = (bf16_t*)take((size_t)HS_ * HS_ * 2);
  bf16_t* Wob  = (bf16_t*)take((size_t)HS_ * HS_ * 2);
  float*  XQ   = (float*) take((size_t)M_ * HS_ * 4);
  float*  XK   = (float*) take((size_t)M_ * HS_ * 4);
  float*  XV   = (float*) take((size_t)M_ * HS_ * 4);
  float*  SIG  = (float*) take((size_t)B_ * NH_ * L_ * 4);
  float*  Obuf = (float*) take((size_t)M_ * HS_ * 4);
  bf16_t* Obf  = (bf16_t*)take((size_t)M_ * HS_ * 2);

  const int nX = M_ * HS_;    // 16,777,216
  const int nW = HS_ * HS_;   //  4,194,304
  cvt_bf16_kernel<<<(nX + 255) / 256, 256, 0, stream>>>(hs, Xbf, nX);
  cvt_bf16_kernel<<<(nW + 255) / 256, 256, 0, stream>>>(qw, Wqb, nW);
  cvt_bf16_kernel<<<(nW + 255) / 256, 256, 0, stream>>>(kw, Wkb, nW);
  cvt_bf16_kernel<<<(nW + 255) / 256, 256, 0, stream>>>(vw, Wvb, nW);
  cvt_bf16_kernel<<<(nW + 255) / 256, 256, 0, stream>>>(ow, Wob, nW);

  dim3 gg(M_ / 128, HS_ / 64), gb(128);
  gemm_bf16_kernel<<<gg, gb, 0, stream>>>(Xbf, Wqb, XQ, HS_, HS_, 1);
  gemm_bf16_kernel<<<gg, gb, 0, stream>>>(Xbf, Wkb, XK, HS_, HS_, 1);
  gemm_bf16_kernel<<<gg, gb, 0, stream>>>(Xbf, Wvb, XV, HS_, HS_, 1);

  lr_sig_kernel<<<M_, 256, 0, stream>>>(hs, lrw, lrb, SIG);

  ttt_scan_kernel<<<B_ * NH_, 128, 0, stream>>>(XQ, XK, XV, SIG, W1, b1, lnw, lnb, tkl, Obuf);

  postln_kernel<<<M_, 256, 0, stream>>>(Obuf, pnw, pnb, Obf);

  gemm_bf16_kernel<<<gg, gb, 0, stream>>>(Obf, Wob, (float*)d_out, HS_, HS_, 0);
}
